// SRNNCellFast_13709535609644
// MI455X (gfx1250) — compile-verified, matrix operand order checked by reading, not run on hardware
//
#include <hip/hip_runtime.h>
#include <hip/hip_bf16.h>
#include <math.h>

// ---------------------------------------------------------------------------
// Problem sizes: B=32, T=1024, IN=512, HID=512, HYP=256
// ---------------------------------------------------------------------------
#define BB    32
#define TT    1024
#define IN_D  512
#define HID_D 512
#define HYP_D 256
#define ROWS  (BB*TT)           // 32768 tokens

typedef __attribute__((ext_vector_type(16))) _Float16 v16h;
typedef __attribute__((ext_vector_type(8)))  float    v8f;
typedef __attribute__((ext_vector_type(4)))  float    f32x4;
typedef __attribute__((ext_vector_type(8)))  _Float16 h8;
typedef __attribute__((ext_vector_type(4)))  _Float16 h4;
typedef __attribute__((ext_vector_type(4)))  int      v4i;

// f16 weight scratch layout inside d_ws (halves)
#define W2H_OFF  0
#define WSH_OFF  (512*512)
#define WMH_OFF  (WSH_OFF + 256*512)
#define WEH_OFF  (WMH_OFF + 256*256)
#define WTOTAL   (WEH_OFF + 512*256)   // 589824 halves = 1.18 MB

// ---------------------------------------------------------------------------
// CDNA5 async global<->LDS (ASYNCcnt) helpers.  Probe round 2 showed the
// builtins exist and take typed pointers:
//   param 1: __attribute__((address_space(1))) int4*   (global)
//   param 2: LDS side (address_space(3))
//   params 3,4: imm offset, imm cpol
// ---------------------------------------------------------------------------
typedef __attribute__((address_space(1))) v4i g_v4i;
typedef __attribute__((address_space(3))) v4i l_v4i;

#if __has_builtin(__builtin_amdgcn_global_load_async_to_lds_b128)
#define HAS_ASYNC_LD 1
#else
#define HAS_ASYNC_LD 0
#endif
#if __has_builtin(__builtin_amdgcn_global_store_async_from_lds_b128)
#define HAS_ASYNC_ST 1
#else
#define HAS_ASYNC_ST 0
#endif

__device__ inline void async_g2l_b128(void* lds, const void* g) {
#if HAS_ASYNC_LD
    __builtin_amdgcn_global_load_async_to_lds_b128(
        (g_v4i*)g, (l_v4i*)lds, 0, 0);
#else
    *(f32x4*)lds = *(const f32x4*)g;          // sync fallback
#endif
}

__device__ inline void async_l2g_b128(void* g, const void* lds) {
#if HAS_ASYNC_ST
    __builtin_amdgcn_global_store_async_from_lds_b128(
        (g_v4i*)g, (l_v4i*)lds, 0, 0);
#else
    *(f32x4*)g = *(const f32x4*)lds;          // sync fallback
#endif
}

#if __has_builtin(__builtin_amdgcn_s_wait_asynccnt)
#define WAIT_ASYNC0() __builtin_amdgcn_s_wait_asynccnt(0)
#else
#define WAIT_ASYNC0() asm volatile("s_wait_asynccnt 0x0" ::: "memory")
#endif

// ---------------------------------------------------------------------------
// WMMA fragment loaders (ISA 7.12.2 16-bit striping):
//   lane L (L<16):  row M=L,    K = [k0..k0+7] ++ [k0+16..k0+23]
//   lane L (L>=16): row M=L-16, K = [k0+8..k0+15] ++ [k0+24..k0+31]
// All GEMMs are act * W^T with W stored [N][K] row-major, so A and B
// fragments are both two contiguous 16B reads from a row.
// ---------------------------------------------------------------------------
__device__ inline v16h load_frag_f16(const _Float16* __restrict__ base, int ldk,
                                     int row, int k0) {
    const _Float16* p = base + (size_t)row * ldk + k0;
    h8 lo = *(const h8*)(p);
    h8 hi = *(const h8*)(p + 16);
    v16h r;
#pragma unroll
    for (int i = 0; i < 8; ++i) { r[i] = lo[i]; r[i + 8] = hi[i]; }
    return r;
}

__device__ inline v16h load_frag_lds(const _Float16* base, int ldk,
                                     int row, int k0) {
    const _Float16* p = base + row * ldk + k0;
    h8 lo = *(const h8*)(p);
    h8 hi = *(const h8*)(p + 16);
    v16h r;
#pragma unroll
    for (int i = 0; i < 8; ++i) { r[i] = lo[i]; r[i + 8] = hi[i]; }
    return r;
}

__device__ inline v16h load_frag_f32cvt(const float* __restrict__ base, int ldk,
                                        int row, int k0) {
    const float* p = base + (size_t)row * ldk + k0;
    f32x4 a = *(const f32x4*)(p);
    f32x4 b = *(const f32x4*)(p + 4);
    f32x4 c = *(const f32x4*)(p + 16);
    f32x4 d = *(const f32x4*)(p + 20);
    v16h r;
#pragma unroll
    for (int i = 0; i < 4; ++i) {
        r[i]      = (_Float16)a[i];
        r[i + 4]  = (_Float16)b[i];
        r[i + 8]  = (_Float16)c[i];
        r[i + 12] = (_Float16)d[i];
    }
    return r;
}

__device__ inline v8f wmma_f16(v16h a, v16h b, v8f c) {
    return __builtin_amdgcn_wmma_f32_16x16x32_f16(false, a, false, b,
                                                  (short)0, c, false, false);
}

// ---------------------------------------------------------------------------
// Kernel 0: one-shot f32 -> f16 weight conversion into d_ws (L2-resident).
// ---------------------------------------------------------------------------
__global__ void cvt_weights_kernel(const float* __restrict__ Ws,
                                   const float* __restrict__ Wm,
                                   const float* __restrict__ We,
                                   const float* __restrict__ W2,
                                   _Float16* __restrict__ wh) {
    int i = blockIdx.x * blockDim.x + threadIdx.x;
    if (i >= WTOTAL) return;
    if (i < WSH_OFF)       wh[i] = (_Float16)W2[i];
    else if (i < WMH_OFF)  wh[i] = (_Float16)Ws[i - WSH_OFF];
    else if (i < WEH_OFF)  wh[i] = (_Float16)Wm[i - WMH_OFF];
    else                   wh[i] = (_Float16)We[i - WEH_OFF];
}

// ---------------------------------------------------------------------------
// Kernel 1: fused MLP.  One wave32 per 32-token tile (M=32): tile-0 x lives
// as A-fragments in VGPRs; tile-1 x is async-DMA'd (f32) into LDS, converted
// to f16 into x2l, and the raw buffer is recycled as h1/h2 staging.  Every
// weight B-fragment load feeds two WMMAs (halves L2 weight traffic vs M=16).
// fc and sig are computed per n-tile back-to-back so no 32x512 gate buffer
// is needed.
// ---------------------------------------------------------------------------
__global__ __launch_bounds__(32)
void mlp_kernel(const float* __restrict__ x,
                const float* __restrict__ bs, const float* __restrict__ bm,
                const float* __restrict__ be, const float* __restrict__ b2,
                const _Float16* __restrict__ W2h, const _Float16* __restrict__ Wsh,
                const _Float16* __restrict__ Wmh, const _Float16* __restrict__ Weh,
                float* __restrict__ out) {
    __shared__ float    xraw[16 * 512];   // 32KB: async-staged raw f32 x (tile1)
    __shared__ _Float16 x2l[16 * 512];    // 16KB: f16 x (tile1)
    _Float16* h1l = (_Float16*)xraw;              // 16KB  [32 x 256]
    _Float16* h2l = ((_Float16*)xraw) + 32 * 256; // 16KB  [32 x 256]

    const int lane  = threadIdx.x;
    const int r0    = blockIdx.x * 32;
    const int mrow  = lane & 15;
    const int khalf = (lane >> 4) * 8;    // 0 or 8

    // 1) kick off async DMA of tile-1 raw x (16 rows x 512 f32 = 32KB)
    {
        const float* g = x + (size_t)(r0 + 16) * IN_D;
#pragma unroll
        for (int q = 0; q < 64; ++q) {
            int off = (q * 32 + lane) * 4;
            async_g2l_b128(xraw + off, g + off);
        }
    }

    // 2) tile-0 x as 16 A-fragments in VGPRs (overlaps the DMA)
    v16h xf[16];
#pragma unroll
    for (int kc = 0; kc < 16; ++kc)
        xf[kc] = load_frag_f32cvt(x + (size_t)r0 * IN_D, IN_D, mrow,
                                  kc * 32 + khalf);

    // 3) wait for DMA, convert raw f32 -> f16 tile in LDS
    WAIT_ASYNC0();
#pragma unroll 8
    for (int q = 0; q < 64; ++q) {
        int off = (q * 32 + lane) * 4;
        f32x4 v = *(const f32x4*)(xraw + off);
        h4 hv;
#pragma unroll
        for (int e = 0; e < 4; ++e) hv[e] = (_Float16)v[e];
        *(h4*)(x2l + off) = hv;
    }

    // ---- h1 = relu(x Ws^T + bs), N=256 (16 n-tiles), K=512 ----------------
    for (int nt = 0; nt < 16; ++nt) {
        v8f a0 = {}, a1 = {};
        for (int kc = 0; kc < 16; ++kc) {
            v16h bf = load_frag_f16(Wsh, 512, nt * 16 + mrow, kc * 32 + khalf);
            a0 = wmma_f16(xf[kc], bf, a0);
            v16h af = load_frag_lds(x2l, 512, mrow, kc * 32 + khalf);
            a1 = wmma_f16(af, bf, a1);
        }
        const int n = nt * 16 + mrow;
        const float bias = bs[n];
#pragma unroll
        for (int j = 0; j < 8; ++j) {
            const int m = j + khalf;
            h1l[m * 256 + n]        = (_Float16)fmaxf(a0[j] + bias, 0.0f);
            h1l[(m + 16) * 256 + n] = (_Float16)fmaxf(a1[j] + bias, 0.0f);
        }
    }

    // ---- h2 = relu(h1 Wm^T + bm), N=256, K=256 ----------------------------
    for (int nt = 0; nt < 16; ++nt) {
        v8f a0 = {}, a1 = {};
        for (int kc = 0; kc < 8; ++kc) {
            v16h bf = load_frag_f16(Wmh, 256, nt * 16 + mrow, kc * 32 + khalf);
            v16h f0 = load_frag_lds(h1l, 256, mrow,      kc * 32 + khalf);
            v16h f1 = load_frag_lds(h1l, 256, mrow + 16, kc * 32 + khalf);
            a0 = wmma_f16(f0, bf, a0);
            a1 = wmma_f16(f1, bf, a1);
        }
        const int n = nt * 16 + mrow;
        const float bias = bm[n];
#pragma unroll
        for (int j = 0; j < 8; ++j) {
            const int m = j + khalf;
            h2l[m * 256 + n]        = (_Float16)fmaxf(a0[j] + bias, 0.0f);
            h2l[(m + 16) * 256 + n] = (_Float16)fmaxf(a1[j] + bias, 0.0f);
        }
    }

    // ---- b = (h2 We^T + be) * sigmoid(x W2^T + b2), N=512 -----------------
    for (int nt = 0; nt < 32; ++nt) {
        v8f f0 = {}, f1 = {};
        for (int kc = 0; kc < 8; ++kc) {
            v16h bf = load_frag_f16(Weh, 256, nt * 16 + mrow, kc * 32 + khalf);
            v16h a0 = load_frag_lds(h2l, 256, mrow,      kc * 32 + khalf);
            v16h a1 = load_frag_lds(h2l, 256, mrow + 16, kc * 32 + khalf);
            f0 = wmma_f16(a0, bf, f0);
            f1 = wmma_f16(a1, bf, f1);
        }
        v8f s0 = {}, s1 = {};
        for (int kc = 0; kc < 16; ++kc) {
            v16h bf = load_frag_f16(W2h, 512, nt * 16 + mrow, kc * 32 + khalf);
            s0 = wmma_f16(xf[kc], bf, s0);
            v16h af = load_frag_lds(x2l, 512, mrow, kc * 32 + khalf);
            s1 = wmma_f16(af, bf, s1);
        }
        const int n = nt * 16 + mrow;
        const float bfc = be[n];
        const float bsg = b2[n];
#pragma unroll
        for (int j = 0; j < 8; ++j) {
            const int m = j + khalf;
            float g0 = 1.0f / (1.0f + __expf(-(s0[j] + bsg)));
            float g1 = 1.0f / (1.0f + __expf(-(s1[j] + bsg)));
            out[(size_t)(r0 + m) * HID_D + n]      = (f0[j] + bfc) * g0;
            out[(size_t)(r0 + 16 + m) * HID_D + n] = (f1[j] + bfc) * g1;
        }
    }
}

// ---------------------------------------------------------------------------
// Kernel 2: recurrence h_t = relu(roll(h_{t-1},1) + b_t), in place on d_out.
// One wave32 per batch row; lane L owns h[L*16..L*16+15] so the roll crosses
// lanes only at one boundary value -> single wrap-around shuffle per step.
// b_t is streamed through a double-buffered 8-step async LDS pipeline and
// outputs leave through an LDS out-buffer with async stores (ASYNCcnt).
// ---------------------------------------------------------------------------
#define CH  8                    // timesteps per chunk
#define CHF (CH * HID_D)         // 4096 floats = 16KB per buffer

__global__ __launch_bounds__(32)
void scan_kernel(const float* __restrict__ hidden,
                 float* __restrict__ out,       // [B,T,512]: b in, outputs out
                 float* __restrict__ fin) {     // [B,512]
    __shared__ float lbuf[2][CHF];   // 2 x 16KB load double buffer
    __shared__ float obuf[CHF];      // 16KB store buffer

    const int b    = blockIdx.x;
    const int lane = threadIdx.x;

    float h[16];
    const float* hp = hidden + (size_t)b * HID_D + lane * 16;
#pragma unroll
    for (int k = 0; k < 16; ++k) h[k] = hp[k];

    float* gbase = out + (size_t)b * TT * HID_D;

    // prime the pipeline: async-load chunk 0
#pragma unroll 4
    for (int q = 0; q < CHF / 128; ++q) {          // 32 b128 per lane
        int off = (q * 32 + lane) * 4;
        async_g2l_b128(&lbuf[0][off], gbase + off);
    }

    for (int c = 0; c < TT / CH; ++c) {
        // waits: chunk-c loads done AND chunk-(c-1) stores drained (obuf free)
        WAIT_ASYNC0();

        if (c + 1 < TT / CH) {
            const float* gnext = gbase + (size_t)(c + 1) * CHF;
#pragma unroll 4
            for (int q = 0; q < CHF / 128; ++q) {
                int off = (q * 32 + lane) * 4;
                async_g2l_b128(&lbuf[(c + 1) & 1][off], gnext + off);
            }
        }
        if (c + 2 < TT / CH)
            __builtin_prefetch(gbase + (size_t)(c + 2) * CHF + lane * 16, 0, 0);

        const float* lb = &lbuf[c & 1][lane * 16];
        float*       ob = &obuf[lane * 16];
#pragma unroll
        for (int s = 0; s < CH; ++s) {
            float bt[16];
            const f32x4* p = (const f32x4*)(lb + s * HID_D);
#pragma unroll
            for (int q = 0; q < 4; ++q) {
                f32x4 v = p[q];
#pragma unroll
                for (int e = 0; e < 4; ++e) bt[q * 4 + e] = v[e];
            }
            // wrap-around rotate-by-1 across the 512-vector (ds_bpermute)
            float prev = __shfl(h[15], lane - 1, 32);
            float nh[16];
            nh[0] = fmaxf(prev + bt[0], 0.0f);
#pragma unroll
            for (int k = 1; k < 16; ++k) nh[k] = fmaxf(h[k - 1] + bt[k], 0.0f);

            f32x4* sp = (f32x4*)(ob + s * HID_D);
#pragma unroll
            for (int q = 0; q < 4; ++q) {
                f32x4 v;
#pragma unroll
                for (int e = 0; e < 4; ++e) v[e] = nh[q * 4 + e];
                sp[q] = v;
            }
#pragma unroll
            for (int k = 0; k < 16; ++k) h[k] = nh[k];
        }

        // async store chunk c back over its input range (disjoint from c+1)
        float* gout = gbase + (size_t)c * CHF;
#pragma unroll 4
        for (int q = 0; q < CHF / 128; ++q) {
            int off = (q * 32 + lane) * 4;
            async_l2g_b128(gout + off, &obuf[off]);
        }
    }
    WAIT_ASYNC0();

    float* fp = fin + (size_t)b * HID_D + lane * 16;
#pragma unroll
    for (int k = 0; k < 16; ++k) fp[k] = h[k];
}

// ---------------------------------------------------------------------------
extern "C" void kernel_launch(void* const* d_in, const int* in_sizes, int n_in,
                              void* d_out, int out_size, void* d_ws, size_t ws_size,
                              hipStream_t stream) {
    const float* x      = (const float*)d_in[0];
    const float* hidden = (const float*)d_in[1];
    const float* Ws     = (const float*)d_in[2];
    const float* bs     = (const float*)d_in[3];
    const float* Wm     = (const float*)d_in[4];
    const float* bm     = (const float*)d_in[5];
    const float* We     = (const float*)d_in[6];
    const float* be     = (const float*)d_in[7];
    const float* W2     = (const float*)d_in[8];
    const float* b2     = (const float*)d_in[9];

    float* out = (float*)d_out;                  // [B,T,512] then [B,512]
    _Float16* wh = (_Float16*)d_ws;

    cvt_weights_kernel<<<(WTOTAL + 255) / 256, 256, 0, stream>>>(Ws, Wm, We, W2, wh);
    mlp_kernel<<<ROWS / 32, 32, 0, stream>>>(x, bs, bm, be, b2,
                                             wh + W2H_OFF, wh + WSH_OFF,
                                             wh + WMH_OFF, wh + WEH_OFF, out);
    scan_kernel<<<BB, 32, 0, stream>>>(hidden, out, out + (size_t)ROWS * HID_D);
}
